// SparseAttention_5317169512736
// MI455X (gfx1250) — matrix operand order repaired and checked by reading, not compiled, not run on hardware
//
#include <hip/hip_runtime.h>
#include <math.h>

// ---------------------------------------------------------------------------
// Problem constants (match reference: B=4,S=2048,E=768,H=12,D=64)
// ---------------------------------------------------------------------------
#define Bb 4
#define Ss 2048
#define Ee 768
#define Hh 12
#define Dd 64
#define LOCAL_H 8
#define HALF_WIN 256      /* window w = clamp(128*sqrt(2048/128),32,512)=512 -> half=256 */
#define NGLOB 204         /* int(2048*0.1) */
#define MASKV -1.0e9f
#define SCALE 0.125f      /* 1/sqrt(64) */

typedef __bf16 bf16_t;
typedef __attribute__((ext_vector_type(16))) __bf16 bf16x16;
typedef __attribute__((ext_vector_type(8)))  __bf16 bf16x8;
typedef __attribute__((ext_vector_type(8)))  float  f32x8;

// ---------------------------------------------------------------------------
// Fragment helpers (CDNA5 wave32 WMMA 16x16x32 bf16 layouts, ISA 7.12.2)
//  A-frag (16x32, row M = lane%16):
//    lanes 0-15 : elems 0..7 = K 0..7,  elems 8..15 = K 16..23
//    lanes16-31 : elems 0..7 = K 8..15, elems 8..15 = K 24..31
//  B-frag (32x16, col N = lane%16):
//    same K order as A-frag: lanes 0-15 : K 0..7, 16..23 ; lanes 16-31 : K 8..15, 24..31
//  C/D f32 (16x16): lane L holds col N=L%16; VGPR r holds row M=r+(L>=16?8:0)
// ---------------------------------------------------------------------------
static __device__ inline bf16x16 load_afrag(const bf16_t* p) {
  bf16x8 lo = *(const bf16x8*)(p);
  bf16x8 hi = *(const bf16x8*)(p + 16);
  bf16x16 r;
#pragma unroll
  for (int i = 0; i < 8; ++i) { r[i] = lo[i]; r[i + 8] = hi[i]; }
  return r;
}

static __device__ inline bf16x16 load_bfrag(const bf16_t* p) {
  bf16x8 lo = *(const bf16x8*)(p);
  bf16x8 hi = *(const bf16x8*)(p + 16);
  bf16x16 r;
#pragma unroll
  for (int i = 0; i < 8; ++i) { r[i] = lo[i]; r[i + 8] = hi[i]; }
  return r;
}

static __device__ inline f32x8 wmma_bf16(bf16x16 a, bf16x16 b, f32x8 c) {
  return __builtin_amdgcn_wmma_f32_16x16x32_bf16(false, a, false, b,
                                                 (short)0, c, false, false);
}

// ---------------------------------------------------------------------------
// Prep kernels
// ---------------------------------------------------------------------------
__global__ void sa_convert_f32_bf16(const float* __restrict__ src,
                                    bf16_t* __restrict__ dst, int n) {
  int i = blockIdx.x * blockDim.x + threadIdx.x;
  if (i < n) dst[i] = (bf16_t)src[i];
}

// w [K,N] row-major f32 -> wT [N,K] row-major bf16
__global__ void sa_transpose_w(const float* __restrict__ w,
                               bf16_t* __restrict__ wT, int K, int N) {
  int i = blockIdx.x * blockDim.x + threadIdx.x;
  if (i < K * N) {
    int k = i / N, n = i - k * N;
    wT[(size_t)n * K + k] = (bf16_t)w[i];
  }
}

// Global-column mask: gpos = int(linspace(0, S-1, NGLOB))
__global__ void sa_gmask(unsigned char* __restrict__ g) {
  for (int i = threadIdx.x; i < Ss; i += blockDim.x) g[i] = 0;
  __syncthreads();
  for (int i = threadIdx.x; i < NGLOB; i += blockDim.x) {
    int pos = (int)((float)i * ((float)(Ss - 1) / (float)(NGLOB - 1)));
    if (pos > Ss - 1) pos = Ss - 1;
    g[pos] = 1;
  }
}

// ---------------------------------------------------------------------------
// Shared GEMM inner loop: 32(M) x 64(N) tile per wave, K = 768, step 32,
// unrolled x2 with two fixed buffer sets (no register rotation copies).
// ---------------------------------------------------------------------------
static __device__ inline void gemm_core(const bf16_t* __restrict__ arow0,
                                        const bf16_t* __restrict__ arow1,
                                        const bf16_t* const* bp,
                                        f32x8 acc[2][4]) {
  bf16x16 aA[2], bA[4], aB[2], bB[4];

  aA[0] = load_afrag(arow0);
  aA[1] = load_afrag(arow1);
#pragma unroll
  for (int t = 0; t < 4; ++t) bA[t] = load_bfrag(bp[t]);

#pragma unroll
  for (int k0 = 0; k0 < Ee; k0 += 64) {
    // prefetch set B at k0+32 (k0+32 <= 736 < 768 always)
    aB[0] = load_afrag(arow0 + k0 + 32);
    aB[1] = load_afrag(arow1 + k0 + 32);
#pragma unroll
    for (int t = 0; t < 4; ++t) bB[t] = load_bfrag(bp[t] + k0 + 32);

#pragma unroll
    for (int t = 0; t < 4; ++t) {
      acc[0][t] = wmma_bf16(aA[0], bA[t], acc[0][t]);
      acc[1][t] = wmma_bf16(aA[1], bA[t], acc[1][t]);
    }

    if (k0 + 64 < Ee) {                // prefetch set A at k0+64
      aA[0] = load_afrag(arow0 + k0 + 64);
      aA[1] = load_afrag(arow1 + k0 + 64);
#pragma unroll
      for (int t = 0; t < 4; ++t) bA[t] = load_bfrag(bp[t] + k0 + 64);
    }

#pragma unroll
    for (int t = 0; t < 4; ++t) {
      acc[0][t] = wmma_bf16(aB[0], bB[t], acc[0][t]);
      acc[1][t] = wmma_bf16(aB[1], bB[t], acc[1][t]);
    }
  }
}

// ---------------------------------------------------------------------------
// GEMM1: qkv = x @ w_attn + b_attn, scattered into Qbf/Kbf[B,H,S,D] and
// Vt[B,H,D,S] (bf16). M = 8192 -> 256 m-blocks, N = 2304 -> 36 n-blocks.
// waves = 9216 -> 1152 blocks.
// ---------------------------------------------------------------------------
__global__ __launch_bounds__(256) void sa_gemm_qkv(
    const bf16_t* __restrict__ xbf, const bf16_t* __restrict__ wT,
    const float* __restrict__ bias, bf16_t* __restrict__ Qb,
    bf16_t* __restrict__ Kb, bf16_t* __restrict__ Vt) {
  const int lane  = threadIdx.x & 31;
  const int wave  = (blockIdx.x << 3) | (threadIdx.x >> 5);
  const int nblk  = wave % 36;
  const int mblk  = wave / 36;
  const int mbase = mblk << 5;
  const int nbase = nblk << 6;
  const int m     = lane & 15;
  const int halfu = lane >> 4;

  f32x8 acc[2][4];
#pragma unroll
  for (int u = 0; u < 2; ++u)
#pragma unroll
    for (int t = 0; t < 4; ++t) acc[u][t] = (f32x8){};

  const bf16_t* arow0 = xbf + (size_t)(mbase + m) * Ee + halfu * 8;
  const bf16_t* arow1 = xbf + (size_t)(mbase + 16 + m) * Ee + halfu * 8;
  const bf16_t* bp[4];
#pragma unroll
  for (int t = 0; t < 4; ++t)
    bp[t] = wT + (size_t)(nbase + t * 16 + m) * Ee + halfu * 8;

  gemm_core(arow0, arow1, bp, acc);

#pragma unroll
  for (int u = 0; u < 2; ++u) {
#pragma unroll
    for (int t = 0; t < 4; ++t) {
#pragma unroll
      for (int r = 0; r < 8; ++r) {
        int mm = mbase + u * 16 + r + halfu * 8;
        int n  = nbase + t * 16 + m;
        float v = acc[u][t][r] + bias[n];
        int which = n / Ee;
        int hd    = n - which * Ee;
        int hh    = hd >> 6;
        int dd    = hd & 63;
        int b     = mm >> 11;
        int s     = mm & 2047;
        size_t bh = (size_t)b * Hh + hh;
        bf16_t bv = (bf16_t)v;
        if (which == 0)      Qb[(bh * Ss + s) * Dd + dd] = bv;
        else if (which == 1) Kb[(bh * Ss + s) * Dd + dd] = bv;
        else                 Vt[(bh * Dd + dd) * Ss + s] = bv;
      }
    }
  }
}

// ---------------------------------------------------------------------------
// Flash attention. One wave per (b, h, 16-row q-tile). S^T = K_tile x Q^T so
// P lands directly in the A-frag layout for P x V (no cross-lane moves).
// V-frag loads and next-chunk K-frag loads are issued before the softmax
// VALU section so exp/mask math hides VMEM latency.
// waves = 4*12*128 = 6144 -> 768 blocks of 256 threads.
// ---------------------------------------------------------------------------
__global__ __launch_bounds__(256) void sa_attn(
    const bf16_t* __restrict__ Qb, const bf16_t* __restrict__ Kb,
    const bf16_t* __restrict__ Vt, const unsigned char* __restrict__ gmask,
    bf16_t* __restrict__ attnbf) {
  const int lane  = threadIdx.x & 31;
  const int wave  = (blockIdx.x << 3) | (threadIdx.x >> 5);
  const int qt    = wave & 127;
  const int bh    = wave >> 7;          // 0..47
  const int h     = bh % Hh;
  const int b     = bh / Hh;
  const int q0    = qt << 4;
  const int m     = lane & 15;
  const int halfu = lane >> 4;
  const bool isLocal = (h < LOCAL_H);
  const int i_row = q0 + m;             // q index of this lane's S^T column

  // Q^T B-fragments (reused across all chunks): d-chunks 0..31 and 32..63
  const bf16_t* qrow = Qb + ((size_t)bh * Ss + (q0 + m)) * Dd;
  bf16x16 qb0 = load_bfrag(qrow + halfu * 8);
  bf16x16 qb1 = load_bfrag(qrow + 32 + halfu * 8);

  f32x8 acc0 = {}, acc1 = {}, acc2 = {}, acc3 = {};
  float rowM = -3.0e38f, rowL = 0.0f;

  int c0 = 0, c1 = (q0 + 15) >> 5;
  if (isLocal) {
    int jlo = q0 - HALF_WIN;
    if (jlo < 0) jlo = 0;
    c0 = jlo >> 5;
  } else {
    c1 = (Ss >> 5) - 1;                 // global heads sweep all columns
  }

  const bf16_t* kbase = Kb + (size_t)bh * Ss * Dd + (size_t)m * Dd + halfu * 8;
  const bf16_t* vbase = Vt + ((size_t)bh * Dd + m) * Ss + halfu * 8;

  // prologue: K A-frags for the first chunk (frag idx = jt*2 + dc)
  bf16x16 ka[4];
  {
    const bf16_t* kr0 = kbase + (size_t)(c0 << 5) * Dd;
    ka[0] = load_afrag(kr0);
    ka[1] = load_afrag(kr0 + 32);
    ka[2] = load_afrag(kr0 + 16 * Dd);
    ka[3] = load_afrag(kr0 + 16 * Dd + 32);
  }

  for (int ch = c0; ch <= c1; ++ch) {
    const int j0 = ch << 5;

    // ---- V B-frags for this chunk: issue loads early ------------------
    const bf16_t* vb = vbase + j0;
    bf16x16 v0 = load_bfrag(vb);
    bf16x16 v1 = load_bfrag(vb + (size_t)16 * Ss);
    bf16x16 v2 = load_bfrag(vb + (size_t)32 * Ss);
    bf16x16 v3 = load_bfrag(vb + (size_t)48 * Ss);

    // ---- S^T tiles: 4 WMMAs -------------------------------------------
    f32x8 z0 = {}, z1 = {};
    f32x8 st0 = wmma_bf16(ka[0], qb0, z0);
    st0 = wmma_bf16(ka[1], qb1, st0);
    f32x8 st1 = wmma_bf16(ka[2], qb0, z1);
    st1 = wmma_bf16(ka[3], qb1, st1);

    // ---- prefetch next chunk's K A-frags ------------------------------
    {
      const int chn = (ch < c1) ? ch + 1 : ch;
      const bf16_t* krn = kbase + (size_t)(chn << 5) * Dd;
      ka[0] = load_afrag(krn);
      ka[1] = load_afrag(krn + 32);
      ka[2] = load_afrag(krn + 16 * Dd);
      ka[3] = load_afrag(krn + 16 * Dd + 32);
    }

    unsigned long long g0 = 0ull, g1 = 0ull;
    if (!isLocal) {
      g0 = *(const unsigned long long*)(gmask + j0 + halfu * 8);
      g1 = *(const unsigned long long*)(gmask + j0 + 16 + halfu * 8);
    }

    // ---- scale + mask, chunk row-max ----------------------------------
    float s0[8], s1[8];
    float mx = -3.0e38f;
#pragma unroll
    for (int r = 0; r < 8; ++r) {
      int j = j0 + halfu * 8 + r;
      bool ok = isLocal ? (j <= i_row && j >= i_row - HALF_WIN)
                        : (j <= i_row || (((g0 >> (8 * r)) & 1ull) != 0ull));
      float v = ok ? st0[r] * SCALE : MASKV;
      s0[r] = v; mx = fmaxf(mx, v);
      int j2 = j + 16;
      bool ok2 = isLocal ? (j2 <= i_row && j2 >= i_row - HALF_WIN)
                         : (j2 <= i_row || (((g1 >> (8 * r)) & 1ull) != 0ull));
      float v2 = ok2 ? st1[r] * SCALE : MASKV;
      s1[r] = v2; mx = fmaxf(mx, v2);
    }
    mx = fmaxf(mx, __shfl_xor(mx, 16, 32));     // combine j-halves of column m
    float Mnew   = fmaxf(rowM, mx);
    float fscale = __expf(rowM - Mnew);

    // ---- probabilities: already in PxV A-frag layout, no shuffles -----
    bf16x16 pa;
    float part = 0.0f;
#pragma unroll
    for (int r = 0; r < 8; ++r) {
      float p0 = __expf(s0[r] - Mnew);
      float p1 = __expf(s1[r] - Mnew);
      part += p0 + p1;
      pa[r]     = (bf16_t)p0;
      pa[r + 8] = (bf16_t)p1;
    }
    part += __shfl_xor(part, 16, 32);
    rowL = rowL * fscale + part;
    rowM = Mnew;

    // ---- rescale accumulators (row of acc elem r is m'=r+halfu*8) -----
#pragma unroll
    for (int r = 0; r < 8; ++r) {
      float fr = __shfl(fscale, r + halfu * 8, 32);
      acc0[r] *= fr; acc1[r] *= fr; acc2[r] *= fr; acc3[r] *= fr;
    }

    // ---- O += P x V : 4 WMMAs -----------------------------------------
    acc0 = wmma_bf16(pa, v0, acc0);
    acc1 = wmma_bf16(pa, v1, acc1);
    acc2 = wmma_bf16(pa, v2, acc2);
    acc3 = wmma_bf16(pa, v3, acc3);
  }

  // ---- finalize: /= rowsum, store attn[b,q,h*64+d] as bf16 ------------
  float linv = 1.0f / rowL;
#pragma unroll
  for (int r = 0; r < 8; ++r) {
    float lr = __shfl(linv, r + halfu * 8, 32);
    int q = q0 + halfu * 8 + r;
    bf16_t* orow = attnbf + ((size_t)b * Ss + q) * Ee + h * Dd + m;
    orow[0]  = (bf16_t)(acc0[r] * lr);
    orow[16] = (bf16_t)(acc1[r] * lr);
    orow[32] = (bf16_t)(acc2[r] * lr);
    orow[48] = (bf16_t)(acc3[r] * lr);
  }
}

// ---------------------------------------------------------------------------
// GEMM2: out = attn @ w_proj + b_proj (fp32 output). 32Mx64N per wave.
// M = 8192 -> 256 m-blocks, N = 768 -> 12 n-blocks. waves = 3072 -> 384 blk.
// ---------------------------------------------------------------------------
__global__ __launch_bounds__(256) void sa_gemm_proj(
    const bf16_t* __restrict__ attnbf, const bf16_t* __restrict__ wT,
    const float* __restrict__ bias, float* __restrict__ out) {
  const int lane  = threadIdx.x & 31;
  const int wave  = (blockIdx.x << 3) | (threadIdx.x >> 5);
  const int nblk  = wave % 12;
  const int mblk  = wave / 12;
  const int mbase = mblk << 5;
  const int nbase = nblk << 6;
  const int m     = lane & 15;
  const int halfu = lane >> 4;

  f32x8 acc[2][4];
#pragma unroll
  for (int u = 0; u < 2; ++u)
#pragma unroll
    for (int t = 0; t < 4; ++t) acc[u][t] = (f32x8){};

  const bf16_t* arow0 = attnbf + (size_t)(mbase + m) * Ee + halfu * 8;
  const bf16_t* arow1 = attnbf + (size_t)(mbase + 16 + m) * Ee + halfu * 8;
  const bf16_t* bp[4];
#pragma unroll
  for (int t = 0; t < 4; ++t)
    bp[t] = wT + (size_t)(nbase + t * 16 + m) * Ee + halfu * 8;

  gemm_core(arow0, arow1, bp, acc);

#pragma unroll
  for (int u = 0; u < 2; ++u) {
#pragma unroll
    for (int t = 0; t < 4; ++t) {
#pragma unroll
      for (int r = 0; r < 8; ++r) {
        int mm = mbase + u * 16 + r + halfu * 8;
        int n  = nbase + t * 16 + m;
        out[(size_t)mm * Ee + n] = acc[u][t][r] + bias[n];
      }
    }
  }
}

// ---------------------------------------------------------------------------
// Host side
// ---------------------------------------------------------------------------
extern "C" void kernel_launch(void* const* d_in, const int* in_sizes, int n_in,
                              void* d_out, int out_size, void* d_ws,
                              size_t ws_size, hipStream_t stream) {
  const float* x      = (const float*)d_in[0];
  const float* w_attn = (const float*)d_in[1];
  const float* b_attn = (const float*)d_in[2];
  const float* w_proj = (const float*)d_in[3];
  const float* b_proj = (const float*)d_in[4];
  float* out = (float*)d_out;

  char* ws = (char*)d_ws;
  size_t off = 0;
  auto carve = [&](size_t bytes) -> char* {
    char* p = ws + off;
    off += (bytes + 255) & ~(size_t)255;
    return p;
  };

  const size_t MROWS = (size_t)Bb * Ss;            // 8192
  bf16_t* xbf    = (bf16_t*)carve(MROWS * Ee * 2);            // 12.6 MB
  bf16_t* wattnT = (bf16_t*)carve((size_t)3 * Ee * Ee * 2);   //  3.5 MB
  bf16_t* wprojT = (bf16_t*)carve((size_t)Ee * Ee * 2);       //  1.2 MB
  bf16_t* Qbf    = (bf16_t*)carve((size_t)Bb * Hh * Ss * Dd * 2);
  bf16_t* Kbf    = (bf16_t*)carve((size_t)Bb * Hh * Ss * Dd * 2);
  bf16_t* Vtbf   = (bf16_t*)carve((size_t)Bb * Hh * Ss * Dd * 2);
  bf16_t* attnbf = (bf16_t*)carve(MROWS * Ee * 2);
  unsigned char* gmask = (unsigned char*)carve(Ss);

  // prep
  {
    int n = (int)(MROWS * Ee);
    sa_convert_f32_bf16<<<(n + 255) / 256, 256, 0, stream>>>(x, xbf, n);
  }
  {
    int n = Ee * 3 * Ee;
    sa_transpose_w<<<(n + 255) / 256, 256, 0, stream>>>(w_attn, wattnT, Ee, 3 * Ee);
  }
  {
    int n = Ee * Ee;
    sa_transpose_w<<<(n + 255) / 256, 256, 0, stream>>>(w_proj, wprojT, Ee, Ee);
  }
  sa_gmask<<<1, 256, 0, stream>>>(gmask);

  // QKV GEMM: 9216 waves / 8 per block
  sa_gemm_qkv<<<1152, 256, 0, stream>>>(xbf, wattnT, b_attn, Qbf, Kbf, Vtbf);

  // attention: 6144 waves
  sa_attn<<<768, 256, 0, stream>>>(Qbf, Kbf, Vtbf, gmask, attnbf);

  // output projection: 3072 waves
  sa_gemm_proj<<<384, 256, 0, stream>>>(attnbf, wprojT, b_proj, out);
}
